// GrapherModule_85375359909952
// MI455X (gfx1250) — matrix operand order repaired and checked by reading, not compiled, not run on hardware
//
#include <hip/hip_runtime.h>
#include <hip/hip_bf16.h>

typedef __attribute__((ext_vector_type(16))) _Float16 v16h;
typedef __attribute__((ext_vector_type(8)))  float    v8f;

union FragU { uint4 q[2]; v16h v; };

// ---------------------------------------------------------------------------
// Problem constants
// ---------------------------------------------------------------------------
#define CB   192          // C
#define NB   3136         // H*W
#define BB   8            // batch
#define TT   25088        // B*N tokens
#define CH   384          // hidden
#define KNN  9

// Workspace layout (bytes, all 256-aligned)
#define OFF_XH   0UL                 // T x 192 f16   (x transposed to token-major)
#define OFF_H    9633792UL           // T x 192 f16   (fc1+bn1 output)
#define OFF_RN   19267584UL          // T f32         (1/row-norm of h)
#define OFF_IDX  19367936UL          // T x 9 int
#define OFF_TCAT 20271104UL          // T x 768 f16   ([t1+gc_b ; t2])
#define OFF_MF   58806272UL          // T x 384 f16   (gelu(bn2(max-pool)))
#define OFF_W1   78073856UL          // 192x192 f16
#define OFF_WG   78147584UL          // 768x192 f16   ([Wa;Wb])
#define OFF_W2   78442496UL          // 192x384 f16

// ---------------------------------------------------------------------------
// Async global->LDS copy of one 32-byte segment (2 x b128, shared IOFFSET
// applies to both the LDS and global address -- CDNA5 ISA).
// Tracked by ASYNCcnt; DMA writes LDS directly, no VGPR round-trip.
// ---------------------------------------------------------------------------
__device__ __forceinline__ void async_copy32(const _Float16* g, _Float16* s) {
  unsigned lds = (unsigned)(uintptr_t)s;   // flat LDS aperture: addr[31:0] = LDS byte offset
  asm volatile(
      "global_load_async_to_lds_b128 %0, %1, off\n\t"
      "global_load_async_to_lds_b128 %0, %1, off offset:16"
      :: "v"(lds), "v"(g) : "memory");
}

// ---------------------------------------------------------------------------
// K0: weight prep: f32 -> f16, build Wa = gcw[:, :C]-gcw[:, C:], Wb = gcw[:, C:]
// ---------------------------------------------------------------------------
__global__ void k_prep_w(const float* __restrict__ fc1_w,
                         const float* __restrict__ gc_w,
                         const float* __restrict__ fc2_w,
                         _Float16* __restrict__ w1,
                         _Float16* __restrict__ wg,
                         _Float16* __restrict__ w2) {
  int i = blockIdx.x * 256 + threadIdx.x;
  if (i < 192 * 192) w1[i] = (_Float16)fc1_w[i];
  if (i < 384 * 192) {
    int o = i / 192, c = i % 192;
    float a = gc_w[o * 384 + c], b = gc_w[o * 384 + 192 + c];
    wg[(long)o * 192 + c]         = (_Float16)(a - b);  // Wa
    wg[(long)(o + 384) * 192 + c] = (_Float16)b;        // Wb
  }
  if (i < 192 * 384) w2[i] = (_Float16)fc2_w[i];
}

// ---------------------------------------------------------------------------
// K1: x (B,C,N) f32 -> xh (token, C) f16, LDS-tiled transpose, fully coalesced
// ---------------------------------------------------------------------------
__global__ void k_pack_x(const float* __restrict__ x, _Float16* __restrict__ xh) {
  __shared__ float tile[32][33];
  int b = blockIdx.z;
  int n0 = blockIdx.x * 32, c0 = blockIdx.y * 32;
  int tx = threadIdx.x, ty = threadIdx.y;
  tile[ty][tx] = x[((long)(b * CB + c0 + ty)) * NB + n0 + tx];
  __syncthreads();
  xh[((long)(b * NB + n0 + ty)) * CB + c0 + tx] = (_Float16)tile[tx][ty];
}

// ---------------------------------------------------------------------------
// WMMA GEMM mainloop: 64x64 tile, 4 waves (each wave: 16 rows x 64 cols).
// K staged through double-buffered LDS in 32-wide chunks via async DMA:
// chunk i+1 streams into LDS while chunk i's 4 WMMAs run. Sync per chunk:
// s_wait_asynccnt 0 (own DMAs landed) + barrier (all waves' DMAs landed).
// Fragment loads use contiguous k = 16*half + s for A and B (not the ISA table order):
// A 16x32 f16 = two b128 LDS reads per lane, B 32x16 f16 likewise.
// sA/sB must each hold 2*64*40 halves.
// ---------------------------------------------------------------------------
__device__ __forceinline__ void gemm_tile_64x64(
    const _Float16* __restrict__ A, const _Float16* __restrict__ W,
    int Kdim, long m0, long n0, int tid,
    _Float16* sA, _Float16* sB, v8f c[4]) {
  const int lane = tid & 31, wid = tid >> 5;
  const int l15 = lane & 15, half = lane >> 4;
  const int row = tid >> 1, seg = tid & 1;
  const _Float16* ga = A + (m0 + row) * (long)Kdim + seg * 16;
  const _Float16* gb = W + (n0 + row) * (long)Kdim + seg * 16;
  _Float16* dA = sA + row * 40 + seg * 16;
  _Float16* dB = sB + row * 40 + seg * 16;
  // prologue: chunk 0 -> buffer 0
  async_copy32(ga, dA);
  async_copy32(gb, dB);
  const int nch = Kdim >> 5;
  for (int i = 0; i < nch; ++i) {
    const int cur = (i & 1) * (64 * 40);
    asm volatile("s_wait_asynccnt 0" ::: "memory");
    __syncthreads();
    if (i + 1 < nch) {                       // overlap next chunk's DMA
      const int nxt = ((i + 1) & 1) * (64 * 40);
      async_copy32(ga + (i + 1) * 32, dA + nxt);
      async_copy32(gb + (i + 1) * 32, dB + nxt);
    }
    const _Float16* cA = sA + cur;
    const _Float16* cB = sB + cur;
    FragU af;
    af.q[0] = *(const uint4*)(cA + (wid * 16 + l15) * 40 + half * 16);
    af.q[1] = *(const uint4*)(cA + (wid * 16 + l15) * 40 + half * 16 + 8);
#pragma unroll
    for (int s = 0; s < 4; ++s) {
      FragU bf;
      const uint4* pb = (const uint4*)(cB + (s * 16 + l15) * 40 + half * 16);
      bf.q[0] = pb[0]; bf.q[1] = pb[1];
      c[s] = __builtin_amdgcn_wmma_f32_16x16x32_f16(false, af.v, false, bf.v,
                                                    (short)0, c[s], false, false);
    }
  }
}

// ---------------------------------------------------------------------------
// K2: h = bn1(xh @ fc1_w^T + fc1_b)  -> f16
// ---------------------------------------------------------------------------
__global__ void __launch_bounds__(128) k_gemm_fc1(
    const _Float16* __restrict__ xh, const _Float16* __restrict__ w1,
    const float* __restrict__ bias, const float* __restrict__ g,
    const float* __restrict__ be, const float* __restrict__ mu,
    const float* __restrict__ va, _Float16* __restrict__ hout) {
  __shared__ __align__(16) _Float16 sA[2 * 64 * 40];
  __shared__ __align__(16) _Float16 sB[2 * 64 * 40];
  v8f c[4] = {};
  long m0 = (long)blockIdx.x * 64, n0 = (long)blockIdx.y * 64;
  gemm_tile_64x64(xh, w1, CB, m0, n0, threadIdx.x, sA, sB, c);
  int lane = threadIdx.x & 31, wid = threadIdx.x >> 5;
  int l15 = lane & 15, half = lane >> 4;
#pragma unroll
  for (int s = 0; s < 4; ++s) {
    int o = (int)n0 + s * 16 + l15;
    float sc = g[o] * rsqrtf(va[o] + 1e-5f);
    float sh = be[o] - mu[o] * sc;
    float bo = bias[o];
#pragma unroll
    for (int r = 0; r < 8; ++r) {
      long t = m0 + wid * 16 + r + half * 8;
      hout[t * CB + o] = (_Float16)((c[s][r] + bo) * sc + sh);
    }
  }
}

// ---------------------------------------------------------------------------
// K2b: rnorm[t] = 1 / max(||h[t]||, 1e-12), one wave32 per token
// ---------------------------------------------------------------------------
__global__ void __launch_bounds__(128) k_rnorm(const _Float16* __restrict__ h,
                                               float* __restrict__ rn) {
  int wid = threadIdx.x >> 5, lane = threadIdx.x & 31;
  long t = (long)blockIdx.x * 4 + wid;
  const _Float16* p = h + t * CB;
  float s = 0.f;
#pragma unroll
  for (int j = 0; j < 6; ++j) {
    float v = (float)p[lane * 6 + j];
    s += v * v;
  }
#pragma unroll
  for (int off = 16; off > 0; off >>= 1) s += __shfl_xor(s, off, 32);
  if (lane == 0) rn[t] = 1.f / fmaxf(sqrtf(s), 1e-12f);
}

// ---------------------------------------------------------------------------
// K3: sim + top-9.  One block per (16-row n-tile, batch). A-fragments for the
// n-tile held in registers; each wave sweeps 49 of the 196 m-tiles, computing
// a 16x16 sim tile with 6 WMMAs (K=192), scaling by rnorm_n*rnorm_m, staging
// the tile in per-wave LDS, and feeding per-lane top-9 insertion lists
// (2 lanes per row). Partial lists merged across the 8 (lane,wave) slots.
// h per batch is 1.2 MB f16 -> resident in the 192 MB L2; B-fragments are
// loaded straight from global (two b128 per lane per chunk).
// Only the index *set* matters downstream (max-pool), so order is free.
// ---------------------------------------------------------------------------
__global__ void __launch_bounds__(128) k_sim_topk(const _Float16* __restrict__ h,
                                                  const float* __restrict__ rn,
                                                  int* __restrict__ idxout) {
  __shared__ __align__(16) _Float16 hA[16 * 200];
  __shared__ float rnAll[NB];
  __shared__ float tileb[4][16][17];
  __shared__ float mv[16][8][9];
  __shared__ int   mi[16][8][9];
  int tid = threadIdx.x, lane = tid & 31, wid = tid >> 5;
  int l15 = lane & 15, half = lane >> 4;
  int b = blockIdx.y;
  long base = (long)b * NB;
  int n0 = blockIdx.x * 16;

  const uint* hsrc = (const uint*)(h + (base + n0) * CB);
  for (int i = tid; i < 16 * 96; i += 128) {
    int row = i / 96, cu = i % 96;
    ((uint*)hA)[row * 100 + cu] = hsrc[row * 96 + cu];
  }
  for (int i = tid; i < NB; i += 128) rnAll[i] = rn[base + i];
  __syncthreads();

  FragU af[6];
#pragma unroll
  for (int kc = 0; kc < 6; ++kc) {
    af[kc].q[0] = *(const uint4*)(hA + l15 * 200 + kc * 32 + half * 16);
    af[kc].q[1] = *(const uint4*)(hA + l15 * 200 + kc * 32 + half * 16 + 8);
  }
  float rnA[8];
#pragma unroll
  for (int r = 0; r < 8; ++r) rnA[r] = rnAll[n0 + r + half * 8];

  float tv[9]; int ti[9];
#pragma unroll
  for (int j = 0; j < 9; ++j) { tv[j] = -3.4e38f; ti[j] = 0; }

  for (int mt = wid; mt < 196; mt += 4) {   // 49 iterations per wave, uniform
    int m0 = mt * 16;
    const _Float16* hm = h + (base + m0 + l15) * CB + half * 16;
    v8f c = {};
#pragma unroll
    for (int kc = 0; kc < 6; ++kc) {        // B frags direct from L2-resident h
      FragU bf;
      const uint4* pb = (const uint4*)(hm + kc * 32);
      bf.q[0] = pb[0]; bf.q[1] = pb[1];
      c = __builtin_amdgcn_wmma_f32_16x16x32_f16(false, af[kc].v, false, bf.v,
                                                 (short)0, c, false, false);
    }
    float rnm = rnAll[m0 + l15];
#pragma unroll
    for (int r = 0; r < 8; ++r)
      tileb[wid][r + half * 8][l15] = c[r] * rnA[r] * rnm;
    // in-wave cross-lane handoff: LDS is in-order per wave; wait for store data
    asm volatile("s_wait_dscnt 0" ::: "memory");
    int row = lane >> 1, ch = lane & 1;
#pragma unroll
    for (int j = 0; j < 8; ++j) {
      float v = tileb[wid][row][ch * 8 + j];
      int mix = m0 + ch * 8 + j;
      if (v > tv[8]) {
        tv[8] = v; ti[8] = mix;
#pragma unroll
        for (int q = 8; q > 0; --q)
          if (tv[q] > tv[q - 1]) {
            float tf = tv[q]; tv[q] = tv[q - 1]; tv[q - 1] = tf;
            int tj = ti[q]; ti[q] = ti[q - 1]; ti[q - 1] = tj;
          }
      }
    }
  }
  {
    int row = lane >> 1, slot = wid * 2 + (lane & 1);
#pragma unroll
    for (int j = 0; j < 9; ++j) { mv[row][slot][j] = tv[j]; mi[row][slot][j] = ti[j]; }
  }
  __syncthreads();
  if (tid < 16) {
    float* pv = &mv[tid][0][0];
    int*   pi = &mi[tid][0][0];
    for (int kk = 0; kk < KNN; ++kk) {
      float best = -3.4e38f; int bidx = 0x7fffffff, bpos = 0;
      for (int q = 0; q < 72; ++q) {
        float v = pv[q];
        if (v > best || (v == best && pi[q] < bidx)) { best = v; bidx = pi[q]; bpos = q; }
      }
      pv[bpos] = -3.4e38f;
      idxout[(base + n0 + tid) * KNN + kk] = bidx;
    }
  }
}

// ---------------------------------------------------------------------------
// K4: tcat = [ Wa@h + gc_b ; Wb@h ]  (768 outputs), f16
// ---------------------------------------------------------------------------
__global__ void __launch_bounds__(128) k_gemm_gc(
    const _Float16* __restrict__ h, const _Float16* __restrict__ wg,
    const float* __restrict__ gcb, _Float16* __restrict__ tcat) {
  __shared__ __align__(16) _Float16 sA[2 * 64 * 40];
  __shared__ __align__(16) _Float16 sB[2 * 64 * 40];
  v8f c[4] = {};
  long m0 = (long)blockIdx.x * 64, n0 = (long)blockIdx.y * 64;
  gemm_tile_64x64(h, wg, CB, m0, n0, threadIdx.x, sA, sB, c);
  int lane = threadIdx.x & 31, wid = threadIdx.x >> 5;
  int l15 = lane & 15, half = lane >> 4;
#pragma unroll
  for (int s = 0; s < 4; ++s) {
    int o = (int)n0 + s * 16 + l15;
    float bo = (o < CH) ? gcb[o] : 0.f;
#pragma unroll
    for (int r = 0; r < 8; ++r) {
      long t = m0 + wid * 16 + r + half * 8;
      tcat[t * 768 + o] = (_Float16)(c[s][r] + bo);
    }
  }
}

// ---------------------------------------------------------------------------
// K5: m = gelu(bn2(t1 + max_k t2[idx_k]))  -> f16. One block per token.
// ---------------------------------------------------------------------------
__global__ void __launch_bounds__(128) k_gather_gelu(
    const _Float16* __restrict__ tcat, const int* __restrict__ idxp,
    const float* __restrict__ g, const float* __restrict__ be,
    const float* __restrict__ mu, const float* __restrict__ va,
    _Float16* __restrict__ mf) {
  long t = blockIdx.x;
  long b = t / NB;
  long rowb[KNN];
#pragma unroll
  for (int k = 0; k < KNN; ++k)
    rowb[k] = (b * NB + idxp[t * KNN + k]) * 768 + CH;
  for (int o = threadIdx.x; o < CH; o += 128) {
    float t1 = (float)tcat[t * 768 + o];
    float mx = -3.4e38f;
#pragma unroll
    for (int k = 0; k < KNN; ++k) mx = fmaxf(mx, (float)tcat[rowb[k] + o]);
    float sc = g[o] * rsqrtf(va[o] + 1e-5f);
    float y = (t1 + mx) * sc + (be[o] - mu[o] * sc);
    y = 0.5f * y * (1.f + erff(y * 0.70710678118f));   // exact gelu
    mf[t * CH + o] = (_Float16)y;
  }
}

// ---------------------------------------------------------------------------
// K6: out = bn3(m @ fc2_w^T + fc2_b) + x, stored directly in (B,C,N) layout
// via an LDS transpose so the n-contiguous stores coalesce/vectorize.
// ---------------------------------------------------------------------------
__global__ void __launch_bounds__(128) k_gemm_fc2(
    const _Float16* __restrict__ mf, const _Float16* __restrict__ w2,
    const float* __restrict__ bias, const float* __restrict__ g,
    const float* __restrict__ be, const float* __restrict__ mu,
    const float* __restrict__ va, const float* __restrict__ x,
    float* __restrict__ out) {
  __shared__ __align__(16) _Float16 sA[2 * 64 * 40];
  __shared__ __align__(16) _Float16 sB[2 * 64 * 40];
  __shared__ __align__(16) float sT[64 * 68];   // pad 68 -> 16B-aligned rows
  v8f c[4] = {};
  long m0 = (long)blockIdx.x * 64, n0 = (long)blockIdx.y * 64;
  gemm_tile_64x64(mf, w2, CH, m0, n0, threadIdx.x, sA, sB, c);
  int tid = threadIdx.x, lane = tid & 31, wid = tid >> 5;
  int l15 = lane & 15, half = lane >> 4;
#pragma unroll
  for (int s = 0; s < 4; ++s)
#pragma unroll
    for (int r = 0; r < 8; ++r)
      sT[(s * 16 + l15) * 68 + wid * 16 + r + half * 8] = c[s][r];
  __syncthreads();
  long b = m0 / NB;               // 3136 = 49*64, tile never crosses a batch
  long nn0 = m0 % NB;
  int orow = tid >> 1, seg = tid & 1;
  int o = (int)n0 + orow;
  float sc = g[o] * rsqrtf(va[o] + 1e-5f);
  float sh = be[o] - mu[o] * sc + bias[o] * sc;
  long base = (b * CB + o) * (long)NB + nn0 + seg * 32;
#pragma unroll
  for (int j = 0; j < 32; ++j) {
    float v = sT[orow * 68 + seg * 32 + j];
    out[base + j] = v * sc + sh + x[base + j];
  }
}

// ---------------------------------------------------------------------------
extern "C" void kernel_launch(void* const* d_in, const int* in_sizes, int n_in,
                              void* d_out, int out_size, void* d_ws, size_t ws_size,
                              hipStream_t stream) {
  const float* x     = (const float*)d_in[0];
  const float* fc1_w = (const float*)d_in[1];
  const float* fc1_b = (const float*)d_in[2];
  const float* bn1_g = (const float*)d_in[3];
  const float* bn1_b = (const float*)d_in[4];
  const float* bn1_m = (const float*)d_in[5];
  const float* bn1_v = (const float*)d_in[6];
  const float* gc_w  = (const float*)d_in[7];
  const float* gc_b  = (const float*)d_in[8];
  const float* bn2_g = (const float*)d_in[9];
  const float* bn2_b = (const float*)d_in[10];
  const float* bn2_m = (const float*)d_in[11];
  const float* bn2_v = (const float*)d_in[12];
  const float* fc2_w = (const float*)d_in[13];
  const float* fc2_b = (const float*)d_in[14];
  const float* bn3_g = (const float*)d_in[15];
  const float* bn3_b = (const float*)d_in[16];
  const float* bn3_m = (const float*)d_in[17];
  const float* bn3_v = (const float*)d_in[18];

  char* ws = (char*)d_ws;
  _Float16* xh   = (_Float16*)(ws + OFF_XH);
  _Float16* hbuf = (_Float16*)(ws + OFF_H);
  float*    rn   = (float*)   (ws + OFF_RN);
  int*      idxb = (int*)     (ws + OFF_IDX);
  _Float16* tcat = (_Float16*)(ws + OFF_TCAT);
  _Float16* mf   = (_Float16*)(ws + OFF_MF);
  _Float16* w1   = (_Float16*)(ws + OFF_W1);
  _Float16* wg   = (_Float16*)(ws + OFF_WG);
  _Float16* w2   = (_Float16*)(ws + OFF_W2);

  k_prep_w<<<288, 256, 0, stream>>>(fc1_w, gc_w, fc2_w, w1, wg, w2);
  k_pack_x<<<dim3(98, 6, 8), dim3(32, 32), 0, stream>>>(x, xh);
  k_gemm_fc1<<<dim3(392, 3), 128, 0, stream>>>(xh, w1, fc1_b, bn1_g, bn1_b,
                                               bn1_m, bn1_v, hbuf);
  k_rnorm<<<6272, 128, 0, stream>>>(hbuf, rn);
  k_sim_topk<<<dim3(196, 8), 128, 0, stream>>>(hbuf, rn, idxb);
  k_gemm_gc<<<dim3(392, 12), 128, 0, stream>>>(hbuf, wg, gc_b, tcat);
  k_gather_gelu<<<25088, 128, 0, stream>>>(tcat, idxb, bn2_g, bn2_b, bn2_m,
                                           bn2_v, mf);
  k_gemm_fc2<<<dim3(392, 3), 128, 0, stream>>>(mf, w2, fc2_b, bn3_g, bn3_b,
                                               bn3_m, bn3_v, x, (float*)d_out);
}
